// KHopSumAggregator_33500744909065
// MI455X (gfx1250) — matrix operand order repaired and checked by reading, not compiled, not run on hardware
//
#include <hip/hip_runtime.h>
#include <hip/hip_bf16.h>
#include <cstdint>

typedef __attribute__((ext_vector_type(16))) __bf16 v16bf;
typedef __attribute__((ext_vector_type(8)))  float  v8f;

#define NNODES 4096
#define NWORDS 128          // 4096 bits / 32
#define BDIM   8
#define FDIM   64
#define MDIM   4
#define KHOPS  3
#define BNF    (BDIM * NNODES * FDIM)   // 2,097,152

// ---------------- bf16 conversion (round-to-nearest-even) ----------------
__device__ __forceinline__ unsigned short f2bf(float f) {
  union { float f; uint32_t u; } v; v.f = f;
  uint32_t u = v.u;
  uint32_t r = u + 0x7FFFu + ((u >> 16) & 1u);
  return (unsigned short)(r >> 16);
}

// ---------------- init: zero adjacency bits, R = identity bitset ----------------
__global__ void init_bits_kernel(uint32_t* __restrict__ adj, uint32_t* __restrict__ R) {
  int i = blockIdx.x * 256 + threadIdx.x;         // over 4096*128 words
  adj[i] = 0u;
  int v  = i >> 7;                                 // row
  int wc = i & 127;                                // word column
  R[i] = (wc == (v >> 5)) ? (1u << (v & 31)) : 0u;
}

// ---------------- scatter edges into bit-adjacency: rowA[u] bit w <=> edge u->w ----
__global__ void build_adj_kernel(const int* __restrict__ ei, uint32_t* __restrict__ adj) {
  int e = blockIdx.x * 256 + threadIdx.x;          // E = 65536
  int u = ei[e];                                   // src
  int w = ei[65536 + e];                           // dst
  atomicOr(&adj[(size_t)u * NWORDS + (w >> 5)], 1u << (w & 31));
}

// ---------------- pows (TRANSPOSED): pw[((m*B+b)*F+f)*N + n] = bf16(|x|^(m+1)) ---------
__global__ void pows_kernel(const float* __restrict__ x, unsigned short* __restrict__ pw) {
  size_t i = (size_t)blockIdx.x * 256 + threadIdx.x;   // over B*F*N
  int n = (int)(i & (NNODES - 1));
  int f = (int)((i >> 12) & (FDIM - 1));
  int b = (int)(i >> 18);
  float ax = fabsf(x[((size_t)b * NNODES + n) * FDIM + f]);
  float p = ax;
  #pragma unroll
  for (int m = 0; m < MDIM; ++m) {
    pw[((size_t)(m * BDIM + b) * FDIM + f) * NNODES + n] = f2bf(p);
    p *= ax;
  }
}

// ---------------- hop: Rn bits update + bf16 expansion (once per hop) ---------------
__global__ void hop_kernel(const uint32_t* __restrict__ Rc,
                           uint32_t* __restrict__ Rn,
                           const uint32_t* __restrict__ adj,
                           unsigned short* __restrict__ Rbf) {
  __shared__ alignas(16) uint32_t rowR[NWORDS];
  int tid = threadIdx.x;
  int v = blockIdx.y;
  if (tid < NWORDS) rowR[tid] = Rc[(size_t)v * NWORDS + tid];
  __syncthreads();

  int u = blockIdx.x * 256 + tid;
  const uint4* a4 = (const uint4*)(adj + (size_t)u * NWORDS);
  const uint4* r4 = (const uint4*)rowR;
  uint4 acc = make_uint4(0u, 0u, 0u, 0u);
  #pragma unroll 8
  for (int j = 0; j < 32; ++j) {
    uint4 a = a4[j];
    uint4 r = r4[j];
    acc.x |= a.x & r.x; acc.y |= a.y & r.y;
    acc.z |= a.z & r.z; acc.w |= a.w & r.w;
  }
  bool inter = ((acc.x | acc.y | acc.z | acc.w) != 0u);
  bool self  = ((rowR[u >> 5] >> (u & 31)) & 1u) != 0u;
  bool reach = inter || self;
  Rbf[(size_t)v * NNODES + u] = reach ? (unsigned short)0x3F80 : (unsigned short)0;
  unsigned long long m = __ballot(reach);
  if ((tid & 31) == 0)
    Rn[(size_t)v * NWORDS + (u >> 5)] = (uint32_t)m;
}

// ---------------- WMMA GEMM: out[b,v,hop,m,:] = Rbf[v,:] @ pows[m,b,:,:] ------------
// grid (16 v-tiles, 32 mb-planes), block 256 (8 waves). Block tile 256(v) x 64(f);
// wave tile 32(v) x 64(f): 2 A-frag x 4 B-frag -> 16 WMMAs per 64-K slab.
// DOUBLE-BUFFERED LDS (90 KB of the WGP's 320 KB): stage slab i+1 while computing
// slab i; one barrier per slab. Staging registers have short live ranges -> no spill.
#define TS 72   // LDS row stride (bf16 elements); 144 B: 16B-aligned, bank stride 36

__global__ void __launch_bounds__(256)
khop_gemm_kernel(const unsigned short* __restrict__ Rbf,
                 const unsigned short* __restrict__ pw,
                 float* __restrict__ out, int hop) {
  __shared__ alignas(16) unsigned short aT[2][256 * TS];  // 2 x 36 KB
  __shared__ alignas(16) unsigned short bT[2][FDIM * TS]; // 2 x  9 KB

  int tid  = threadIdx.x;
  int wave = tid >> 5;
  int lane = tid & 31;
  int v0   = blockIdx.x * 256;
  int mb   = blockIdx.y;                       // mb = m*B + b
  const unsigned short* P = pw + (size_t)mb * ((size_t)FDIM * NNODES); // [f][n]

  v8f acc[2][4] = {};

  // staging slot per thread: rows srow + j*32, uint4 index sq
  int srow = tid >> 3;                         // 0..31
  int sq   = tid & 7;                          // 0..7

  int arow = wave * 32 + (lane & 15);          // fragment base row (A) per lane
  int akb  = (lane >> 4) * 8;                  // A K-sub-offset per half-wave
  int bcol = lane & 15;                        // fragment col (B) per lane
  int bkb  = (lane >> 4) * 8;                  // B K-sub-offset per half-wave

  auto stage = [&](int n0, int buf) {
    #pragma unroll
    for (int j = 0; j < 8; ++j)
      ((uint4*)&aT[buf][(srow + j * 32) * TS])[sq] =
          ((const uint4*)&Rbf[(size_t)(v0 + srow + j * 32) * NNODES + n0])[sq];
    #pragma unroll
    for (int j = 0; j < 2; ++j)
      ((uint4*)&bT[buf][(srow + j * 32) * TS])[sq] =
          ((const uint4*)&P[(size_t)(srow + j * 32) * NNODES + n0])[sq];
  };

  stage(0, 0);
  __syncthreads();

  int buf = 0;
  for (int n0 = 0; n0 < NNODES; n0 += 64) {
    // --- stage next slab into the other buffer (overlaps with WMMAs below)
    if (n0 + 64 < NNODES) stage(n0 + 64, buf ^ 1);

    // --- compute: 2 K-sub-slabs of 32; per slab: 2 A-frag, 4 B-frag, 8 WMMAs
    #pragma unroll
    for (int ks = 0; ks < 2; ++ks) {
      union { uint4 q[2]; v16bf v; } afr[2];
      #pragma unroll
      for (int r = 0; r < 2; ++r) {
        afr[r].q[0] = *(const uint4*)&aT[buf][(arow + r * 16) * TS + ks * 32 + akb];
        afr[r].q[1] = *(const uint4*)&aT[buf][(arow + r * 16) * TS + ks * 32 + akb + 16];
      }
      union { uint4 q[2]; v16bf v; } bfr[4];
      #pragma unroll
      for (int t = 0; t < 4; ++t) {
        int f = t * 16 + bcol;
        bfr[t].q[0] = *(const uint4*)&bT[buf][f * TS + ks * 32 + bkb];
        bfr[t].q[1] = *(const uint4*)&bT[buf][f * TS + ks * 32 + bkb + 16];
      }
      #pragma unroll
      for (int r = 0; r < 2; ++r)
        #pragma unroll
        for (int t = 0; t < 4; ++t)
          acc[r][t] = __builtin_amdgcn_wmma_f32_16x16x32_bf16(
              false, afr[r].v, false, bfr[t].v, (short)0, acc[r][t], false, false);
    }
    __syncthreads();   // staging of buf^1 done; safe to read next iter & rewrite buf
    buf ^= 1;
  }

  // --- epilogue: C layout: col = lane&15; rows = (lane>>4)*8 + rr
  int m = mb >> 3;           // mb = m*B + b
  int b = mb & 7;
  int mrow0 = (lane >> 4) * 8;
  #pragma unroll
  for (int r = 0; r < 2; ++r) {
    #pragma unroll
    for (int t = 0; t < 4; ++t) {
      #pragma unroll
      for (int rr = 0; rr < 8; ++rr) {
        int v = v0 + wave * 32 + r * 16 + mrow0 + rr;
        int f = t * 16 + bcol;
        size_t o = ((((size_t)b * NNODES + v) * KHOPS + hop) * MDIM + m) * FDIM + f;
        out[o] = acc[r][t][rr];
      }
    }
  }
}

extern "C" void kernel_launch(void* const* d_in, const int* in_sizes, int n_in,
                              void* d_out, int out_size, void* d_ws, size_t ws_size,
                              hipStream_t stream) {
  const float* x  = (const float*)d_in[0];
  const int*   ei = (const int*)d_in[1];
  float* out = (float*)d_out;

  uint8_t* ws = (uint8_t*)d_ws;
  uint32_t* adj = (uint32_t*)(ws);                               // 2 MB
  uint32_t* R0  = (uint32_t*)(ws + ((size_t)2 << 20));           // 2 MB
  uint32_t* R1  = (uint32_t*)(ws + ((size_t)4 << 20));           // 2 MB
  unsigned short* pw  = (unsigned short*)(ws + ((size_t)6 << 20));  // 16 MB, [mb][f][n]
  unsigned short* Rbf = (unsigned short*)(ws + ((size_t)22 << 20)); // 32 MB, [v][n]

  init_bits_kernel<<<(NNODES * NWORDS) / 256, 256, 0, stream>>>(adj, R0);
  build_adj_kernel<<<65536 / 256, 256, 0, stream>>>(ei, adj);
  pows_kernel<<<BNF / 256, 256, 0, stream>>>(x, pw);

  uint32_t* rc = R0;
  uint32_t* rn = R1;
  for (int k = 0; k < KHOPS; ++k) {
    hop_kernel<<<dim3(NNODES / 256, NNODES), 256, 0, stream>>>(rc, rn, adj, Rbf);
    khop_gemm_kernel<<<dim3(NNODES / 256, MDIM * BDIM), 256, 0, stream>>>(Rbf, pw, out, k);
    uint32_t* t = rc; rc = rn; rn = t;
  }
}
